// GCN_29712583754009
// MI455X (gfx1250) — compile-verified, matrix operand order checked by reading, not run on hardware
//
#include <hip/hip_runtime.h>

// ---------------------------------------------------------------------------
// GCN forward (2x GCNConv(128->128) + ReLU, head 128->16) for gfx1250.
//  - GEMMs: v_wmma_f32_16x16x32_bf16, f32 accumulate. Weights are pre-packed
//    once into exact WMMA B-fragment order (bf16, lane-contiguous) so the hot
//    K-loop is just b128 loads + wmma.
//  - Aggregation: wave-per-edge gather + global_atomic_add_f32 scatter on an
//    L2-resident (51 MB) buffer.
// ---------------------------------------------------------------------------

typedef __attribute__((ext_vector_type(16))) __bf16 v16bf;
typedef __attribute__((ext_vector_type(8)))  float  v8f;

#define GCN_N    100000
#define GCN_DIM  128

// Per ISA 16-bit A/B fragment layout (wave32):
//   lane = half*16 + (m or n);  VGPR v (0..7) holds element pair (2v, 2v+1)
//   k(pair v) = [v<4 ? 2v : 16 + 2(v-4)] + 8*half   (+ ks*32)
__device__ __forceinline__ int frag_k(int v, int half) {
    return ((v < 4) ? (2 * v) : (16 + 2 * (v - 4))) + 8 * half;
}

// ---------------- degree / normalization ------------------------------------

__global__ void k_deg_init(float* __restrict__ deg, int n) {
    int i = blockIdx.x * blockDim.x + threadIdx.x;
    if (i < n) deg[i] = 1.0f;  // self-loop
}

__global__ void k_deg_edges(const int* __restrict__ dst, float* __restrict__ deg, int E) {
    int e = blockIdx.x * blockDim.x + threadIdx.x;
    if (e < E) atomicAdd(deg + dst[e], 1.0f);
}

__global__ void k_rsqrt_inplace(float* __restrict__ d, int n) {
    int i = blockIdx.x * blockDim.x + threadIdx.x;
    if (i < n) d[i] = rsqrtf(d[i]);  // deg >= 1 always (self-loops)
}

// ---------------- weight pre-pack into WMMA B-fragment order ----------------
// out fragment f = nt*4 + ks; storage: out[(f*32 + lane)*16 + j] (bf16),
// so a wave loads one fragment as a single contiguous 32B chunk per lane.
__global__ void k_pack_b(const float* __restrict__ W, __bf16* __restrict__ out,
                         int Ncols) {
    int idx = blockIdx.x * blockDim.x + threadIdx.x;  // one (fragment, lane)
    int total = (Ncols / 16) * 4 * 32;
    if (idx >= total) return;
    int lane = idx & 31;
    int ks   = (idx >> 5) & 3;
    int nt   = idx >> 7;
    int half = lane >> 4;
    int n    = nt * 16 + (lane & 15);
    __bf16* o = out + (size_t)idx * 16;
#pragma unroll
    for (int j = 0; j < 16; ++j) {
        int k = ks * 32 + frag_k(j >> 1, half) + (j & 1);
        o[j] = (__bf16)W[(size_t)k * Ncols + n];
    }
}

// ---------------- WMMA GEMM: [M,128] x packed[128,128] -> [M,128] -----------
// One wave owns a 16-row strip; A fragments built once, B fragments are
// single v16bf loads from the packed weights.
__global__ void k_gemm_bf16_n128(const float* __restrict__ A,
                                 const __bf16* __restrict__ Bp,
                                 float* __restrict__ C, int M) {
    const int K = 128, Nc = 128;
    int wave = blockIdx.x * (blockDim.x >> 5) + (threadIdx.x >> 5);
    int r0 = wave * 16;
    if (r0 >= M) return;                 // wave-uniform: EXEC stays all-ones
    int lane = threadIdx.x & 31;
    int half = lane >> 4;
    int r    = r0 + (lane & 15);
    int nl   = lane & 15;

    const float* ar = A + (size_t)r * K;
    v16bf afrag[4];
#pragma unroll
    for (int ks = 0; ks < 4; ++ks) {
        v16bf a;
#pragma unroll
        for (int v = 0; v < 8; ++v) {
            int k = ks * 32 + frag_k(v, half);
            float2 t = *(const float2*)(ar + k);
            a[2 * v]     = (__bf16)t.x;
            a[2 * v + 1] = (__bf16)t.y;
        }
        afrag[ks] = a;
    }

    const v16bf* bfr = (const v16bf*)Bp;
#pragma unroll
    for (int nt = 0; nt < 8; ++nt) {
        v8f acc = {};
#pragma unroll
        for (int ks = 0; ks < 4; ++ks) {
            v16bf b = bfr[(size_t)(nt * 4 + ks) * 32 + lane];
            acc = __builtin_amdgcn_wmma_f32_16x16x32_bf16(
                false, afrag[ks], false, b, (short)0, acc, false, false);
        }
        // D layout: VGPR v -> row (v + 8*half), col = lane&15
#pragma unroll
        for (int v = 0; v < 8; ++v)
            C[(size_t)(r0 + half * 8 + v) * Nc + nt * 16 + nl] = acc[v];
    }
}

// ---------------- WMMA GEMM head: [M,128] x packed[128,16] + bias -----------

__global__ void k_gemm_bf16_n16(const float* __restrict__ A,
                                const __bf16* __restrict__ Bp,
                                const float* __restrict__ bias,
                                float* __restrict__ C, int M) {
    const int K = 128, Nc = 16;
    int wave = blockIdx.x * (blockDim.x >> 5) + (threadIdx.x >> 5);
    int r0 = wave * 16;
    if (r0 >= M) return;
    int lane = threadIdx.x & 31;
    int half = lane >> 4;
    int r    = r0 + (lane & 15);
    int nl   = lane & 15;

    const float* ar = A + (size_t)r * K;
    const v16bf* bfr = (const v16bf*)Bp;
    float bn = bias[nl];

    v8f acc = {};
#pragma unroll
    for (int ks = 0; ks < 4; ++ks) {
        v16bf a;
#pragma unroll
        for (int v = 0; v < 8; ++v) {
            int k = ks * 32 + frag_k(v, half);
            float2 t = *(const float2*)(ar + k);
            a[2 * v]     = (__bf16)t.x;
            a[2 * v + 1] = (__bf16)t.y;
        }
        v16bf b = bfr[(size_t)ks * 32 + lane];
        acc = __builtin_amdgcn_wmma_f32_16x16x32_bf16(
            false, a, false, b, (short)0, acc, false, false);
    }
#pragma unroll
    for (int v = 0; v < 8; ++v)
        C[(size_t)(r0 + half * 8 + v) * Nc + nl] = acc[v] + bn;
}

// ---------------- aggregation ------------------------------------------------
// out[i] = h[i] * dinv[i]^2   (self-loop term, also initializes out)
__global__ void k_self_init(const float* __restrict__ h,
                            const float* __restrict__ dinv,
                            float* __restrict__ out, int n) {
    int idx = blockIdx.x * blockDim.x + threadIdx.x;  // one float4 per thread
    int total = n * (GCN_DIM / 4);
    if (idx >= total) return;
    int node = idx >> 5;  // 32 float4 per row
    float di = dinv[node];
    float s = di * di;
    float4 v = ((const float4*)h)[idx];
    v.x *= s; v.y *= s; v.z *= s; v.w *= s;
    ((float4*)out)[idx] = v;
}

// agg[dst] += h[src] * dinv[src]*dinv[dst]  -- one wave per edge, float4/lane
__global__ void k_edge_agg(const float* __restrict__ h,
                           const int* __restrict__ src,
                           const int* __restrict__ dst,
                           const float* __restrict__ dinv,
                           float* __restrict__ agg, int E) {
    int e = (blockIdx.x * blockDim.x + threadIdx.x) >> 5;
    if (e >= E) return;
    int lane = threadIdx.x & 31;
    int s = src[e], d = dst[e];
    float norm = dinv[s] * dinv[d];
    float4 v = ((const float4*)(h + (size_t)s * GCN_DIM))[lane];
    float* ad = agg + (size_t)d * GCN_DIM + lane * 4;
    atomicAdd(ad + 0, v.x * norm);
    atomicAdd(ad + 1, v.y * norm);
    atomicAdd(ad + 2, v.z * norm);
    atomicAdd(ad + 3, v.w * norm);
}

// out = relu(in + bias)
__global__ void k_bias_relu(const float* __restrict__ in,
                            const float* __restrict__ bias,
                            float* __restrict__ out, int n) {
    int idx = blockIdx.x * blockDim.x + threadIdx.x;  // one float4 per thread
    int total = n * (GCN_DIM / 4);
    if (idx >= total) return;
    int c4 = idx & 31;
    float4 b = ((const float4*)bias)[c4];
    float4 v = ((const float4*)in)[idx];
    v.x = fmaxf(v.x + b.x, 0.0f);
    v.y = fmaxf(v.y + b.y, 0.0f);
    v.z = fmaxf(v.z + b.z, 0.0f);
    v.w = fmaxf(v.w + b.w, 0.0f);
    ((float4*)out)[idx] = v;
}

// ---------------------------------------------------------------------------

extern "C" void kernel_launch(void* const* d_in, const int* in_sizes, int n_in,
                              void* d_out, int out_size, void* d_ws, size_t ws_size,
                              hipStream_t stream) {
    (void)n_in; (void)out_size; (void)ws_size;
    const float* x   = (const float*)d_in[0];
    const float* W1  = (const float*)d_in[1];
    const float* b1  = (const float*)d_in[2];
    const float* W2  = (const float*)d_in[3];
    const float* b2  = (const float*)d_in[4];
    const float* Wh  = (const float*)d_in[5];
    const float* bh  = (const float*)d_in[6];
    const int*   ei  = (const int*)d_in[7];
    float*       out = (float*)d_out;

    const int N = GCN_N;
    const int E = in_sizes[7] / 2;     // edge_index is [2, E] flat
    const int* src = ei;
    const int* dst = ei + E;

    // workspace layout (floats): dinv | bufA | bufB | packed weights (bf16)
    float* dinv = (float*)d_ws;
    float* bufA = dinv + 100096;                   // padded past N
    float* bufB = bufA + (size_t)N * GCN_DIM;
    __bf16* W1p = (__bf16*)(bufB + (size_t)N * GCN_DIM);  // 32B aligned
    __bf16* W2p = W1p + 8 * 4 * 32 * 16;           // 16384 bf16 each
    __bf16* Whp = W2p + 8 * 4 * 32 * 16;           // + 2048 bf16

    const int TB = 256;
    int blkN   = (N + TB - 1) / TB;
    int blkE   = (E + TB - 1) / TB;
    int blkEl  = (N * (GCN_DIM / 4) + TB - 1) / TB;   // elementwise float4
    int blkEw  = (E + 7) / 8;                         // one wave per edge
    int tiles  = N / 16;                              // 6250, exact
    int blkG   = (tiles + 7) / 8;                     // 8 waves per block

    // one-time packs (cheap; deterministic each call)
    k_pack_b<<<4, TB, 0, stream>>>(W1, W1p, 128);     // 1024 threads
    k_pack_b<<<4, TB, 0, stream>>>(W2, W2p, 128);
    k_pack_b<<<1, TB, 0, stream>>>(Wh, Whp, 16);      // 128 threads

    // normalization: dinv = rsqrt(1 + in-degree)
    k_deg_init<<<blkN, TB, 0, stream>>>(dinv, N);
    k_deg_edges<<<blkE, TB, 0, stream>>>(dst, dinv, E);
    k_rsqrt_inplace<<<blkN, TB, 0, stream>>>(dinv, N);

    // layer 1: h = relu(Agg(x @ W1) + b1)
    k_gemm_bf16_n128<<<blkG, TB, 0, stream>>>(x, W1p, bufA, N);
    k_self_init<<<blkEl, TB, 0, stream>>>(bufA, dinv, bufB, N);
    k_edge_agg<<<blkEw, TB, 0, stream>>>(bufA, src, dst, dinv, bufB, E);
    k_bias_relu<<<blkEl, TB, 0, stream>>>(bufB, b1, bufA, N);

    // layer 2: h = relu(Agg(h @ W2) + b2)
    k_gemm_bf16_n128<<<blkG, TB, 0, stream>>>(bufA, W2p, bufB, N);
    k_self_init<<<blkEl, TB, 0, stream>>>(bufB, dinv, bufA, N);
    k_edge_agg<<<blkEw, TB, 0, stream>>>(bufB, src, dst, dinv, bufA, E);
    k_bias_relu<<<blkEl, TB, 0, stream>>>(bufA, b2, bufB, N);

    // head: out = h @ Wh + bh
    k_gemm_bf16_n16<<<blkG, TB, 0, stream>>>(bufB, Whp, bh, out, N);
}
